// CFGGNN_V2_78477642432716
// MI455X (gfx1250) — matrix operand order repaired and checked by reading, not compiled, not run on hardware
//
#include <hip/hip_runtime.h>
#include <hip/hip_bf16.h>
#include <math.h>

typedef _Float16 v16h __attribute__((ext_vector_type(16)));
typedef float    v8f  __attribute__((ext_vector_type(8)));

#define HDIM 128
#define EPSV 1e-5f

// ---------------------------------------------------------------------------
// Utility fills
// ---------------------------------------------------------------------------
__global__ void fill_f32(float* __restrict__ p, float v, int n) {
    int i = blockIdx.x * blockDim.x + threadIdx.x;
    if (i < n) p[i] = v;
}

// Convert W[k][n] (fp32, 128x128) -> Wt[n][k] (fp16, transposed)
__global__ void wt_convert(const float* __restrict__ W, _Float16* __restrict__ Wt) {
    int idx = blockIdx.x * blockDim.x + threadIdx.x;   // 16384 threads
    if (idx >= HDIM * HDIM) return;
    int k = idx >> 7, n = idx & 127;
    Wt[n * HDIM + k] = (_Float16)W[idx];
}

// deg[dst]++ per edge (deg pre-initialized to 1.0 for self-loop)
__global__ void deg_count(const int* __restrict__ dst, float* __restrict__ deg, int E) {
    int i = blockIdx.x * blockDim.x + threadIdx.x;
    if (i < E) unsafeAtomicAdd(&deg[dst[i]], 1.0f);
}

__global__ void dinv_kernel(const float* __restrict__ deg, float* __restrict__ dinv, int n) {
    int i = blockIdx.x * blockDim.x + threadIdx.x;
    if (i < n) dinv[i] = rsqrtf(deg[i]);   // deg >= 1 always (self-loop)
}

// ---------------------------------------------------------------------------
// WMMA GEMM: C[N,128] = A[N,128] (fp32, cvt->f16) @ Wt^T (Wt is [n][k] fp16)
// One wave computes a 16-row stripe (16 x 128), 8 N-tiles x 4 K-steps.
// Fragment packing follows CDNA5 ISA 7.12.2 layouts (16-bit A 16x32, B 32x16,
// 32-bit C/D 16x16). Tail guard is wave-uniform (single scalar branch).
// ---------------------------------------------------------------------------
__global__ void gemm128_wmma(const float* __restrict__ A, const _Float16* __restrict__ Wt,
                             float* __restrict__ C, int nRowTiles, int N) {
    int wave = (blockIdx.x * blockDim.x + threadIdx.x) >> 5;
    int lane = threadIdx.x & 31;
    if (wave >= nRowTiles) return;
    int rowBase = wave << 4;
    int mlo  = lane & 15;
    bool hi  = lane >= 16;
    bool fullTile = (rowBase + 16 <= N);   // wave-uniform

    int arow_i = rowBase + mlo;
    if (arow_i >= N) arow_i = N - 1;       // tail guard (loads only)
    const float* arow = A + (size_t)arow_i * HDIM;

    // A fragments for K = 0..127 (4 frags of 16x32)
    v16h afrag[4];
#pragma unroll
    for (int kk = 0; kk < 4; ++kk) {
        int k0 = kk * 32 + (hi ? 8 : 0);
#pragma unroll
        for (int j = 0; j < 8; ++j) {
            afrag[kk][j]     = (_Float16)arow[k0 + j];        // K = k0..k0+7
            afrag[kk][8 + j] = (_Float16)arow[k0 + 16 + j];   // K = k0+16..k0+23
        }
    }

    int kboff = hi ? 8 : 0;
    // base pointer for this lane's column stores: row (rowBase + hi?8:0), col ncol
    float* crow0 = C + (size_t)(rowBase + (hi ? 8 : 0)) * HDIM + mlo;

#pragma unroll
    for (int nt = 0; nt < 8; ++nt) {
        int ncol = nt * 16;
        const _Float16* wcol = Wt + (size_t)(ncol + mlo) * HDIM;
        v8f acc = {};
#pragma unroll
        for (int kk = 0; kk < 4; ++kk) {
            const _Float16* bp = wcol + kk * 32 + kboff;      // K = k0..k0+7, k0+16..k0+23
            v16h b;
#pragma unroll
            for (int j = 0; j < 8; ++j) { b[j] = bp[j]; b[8 + j] = bp[16 + j]; }
            acc = __builtin_amdgcn_wmma_f32_16x16x32_f16(
                false, afrag[kk], false, b, (short)0, acc, false, false);
        }
        // C/D layout: lanes 0-15 -> M=v, lanes 16-31 -> M=v+8 ; N = lane&15
        float* cp = crow0 + ncol;
        if (fullTile) {
#pragma unroll
            for (int v = 0; v < 8; ++v) cp[(size_t)v * HDIM] = acc[v];
        } else {
#pragma unroll
            for (int v = 0; v < 8; ++v) {
                int m = hi ? (v + 8) : v;
                if (rowBase + m < N) cp[(size_t)v * HDIM] = acc[v];
            }
        }
    }
}

// ---------------------------------------------------------------------------
// Edge scatter: agg[dst] += hw[src] * dinv[src]*dinv[dst]  (one wave per edge)
// ---------------------------------------------------------------------------
__global__ void edge_scatter(const float* __restrict__ hw, const float* __restrict__ dinv,
                             const int* __restrict__ src, const int* __restrict__ dst,
                             float* __restrict__ agg, int E) {
    int e = (blockIdx.x * blockDim.x + threadIdx.x) >> 5;
    int lane = threadIdx.x & 31;
    if (e >= E) return;
    int s = src[e], d = dst[e];
    float w = dinv[s] * dinv[d];
    const float4 hv = *(const float4*)(hw + (size_t)s * HDIM + lane * 4);
    float* o = agg + (size_t)d * HDIM + lane * 4;
    unsafeAtomicAdd(o + 0, hv.x * w);
    unsafeAtomicAdd(o + 1, hv.y * w);
    unsafeAtomicAdd(o + 2, hv.z * w);
    unsafeAtomicAdd(o + 3, hv.w * w);
}

// ---------------------------------------------------------------------------
// Self-loop add + per-channel BN statistics (partial sums per block).
// blockDim = 128 (thread = channel), 64 rows per block.
// ---------------------------------------------------------------------------
__global__ void selfloop_stats(const float* __restrict__ hw, const float* __restrict__ dinv,
                               float* __restrict__ agg, float* __restrict__ sum,
                               float* __restrict__ sumsq, int N) {
    int c = threadIdx.x;
    int row0 = blockIdx.x * 64;
    float s = 0.f, q = 0.f;
    for (int r = 0; r < 64; ++r) {
        int row = row0 + r;
        if (row >= N) break;
        float di = dinv[row];
        size_t idx = (size_t)row * HDIM + c;
        float v = agg[idx] + hw[idx] * (di * di);   // self-loop norm = 1/deg
        agg[idx] = v;
        s += v; q += v * v;
    }
    unsafeAtomicAdd(&sum[c], s);
    unsafeAtomicAdd(&sumsq[c], q);
}

// ---------------------------------------------------------------------------
// BN normalize + optional ReLU + optional residual. (GCN bias cancels in BN.)
// ---------------------------------------------------------------------------
__global__ void bn_apply(const float* __restrict__ agg, const float* __restrict__ sum,
                         const float* __restrict__ sumsq, const float* __restrict__ g,
                         const float* __restrict__ be, const float* __restrict__ resid,
                         float* __restrict__ out, int N, int doRelu) {
    int idx = blockIdx.x * blockDim.x + threadIdx.x;
    if (idx >= N * HDIM) return;
    int c = idx & 127;
    float invn = 1.0f / (float)N;
    float m   = sum[c] * invn;
    float var = sumsq[c] * invn - m * m;
    float rstd = rsqrtf(var + EPSV);
    float v = (agg[idx] - m) * rstd * g[c] + be[c];
    if (doRelu) v = fmaxf(v, 0.0f);
    if (resid)  v += resid[idx];
    out[idx] = v;
}

// ---------------------------------------------------------------------------
// Attention pooling: one wave per node. wave32 shuffle reduction for the dot.
// ---------------------------------------------------------------------------
__global__ void pool_attn(const float* __restrict__ h, const float* __restrict__ aw,
                          const float* __restrict__ ab, const int* __restrict__ batch,
                          float* __restrict__ gout, int N) {
    int node = (blockIdx.x * blockDim.x + threadIdx.x) >> 5;
    int lane = threadIdx.x & 31;
    if (node >= N) return;
    const float4 hv = *(const float4*)(h + (size_t)node * HDIM + lane * 4);
    const float4 wv = *(const float4*)(aw + lane * 4);
    float d = hv.x * wv.x + hv.y * wv.y + hv.z * wv.z + hv.w * wv.w;
#pragma unroll
    for (int off = 16; off > 0; off >>= 1) d += __shfl_xor(d, off, 32);
    float a = 1.0f / (1.0f + __expf(-(d + ab[0])));
    int b = batch[node];
    float* o = gout + (size_t)b * HDIM + lane * 4;
    unsafeAtomicAdd(o + 0, hv.x * a);
    unsafeAtomicAdd(o + 1, hv.y * a);
    unsafeAtomicAdd(o + 2, hv.z * a);
    unsafeAtomicAdd(o + 3, hv.w * a);
}

// ---------------------------------------------------------------------------
// Host-side orchestration
// ---------------------------------------------------------------------------
static inline int cdiv(long long a, long long b) { return (int)((a + b - 1) / b); }

extern "C" void kernel_launch(void* const* d_in, const int* in_sizes, int n_in,
                              void* d_out, int out_size, void* d_ws, size_t ws_size,
                              hipStream_t stream) {
    const float* x      = (const float*)d_in[0];
    const int*   ei     = (const int*)  d_in[1];
    const int*   batch  = (const int*)  d_in[2];
    const float* W_in   = (const float*)d_in[3];
    const float* g_in   = (const float*)d_in[5];
    const float* be_in  = (const float*)d_in[6];
    const float* W_m0   = (const float*)d_in[7];
    const float* W_m1   = (const float*)d_in[8];
    const float* g_m0   = (const float*)d_in[11];
    const float* g_m1   = (const float*)d_in[12];
    const float* be_m0  = (const float*)d_in[13];
    const float* be_m1  = (const float*)d_in[14];
    const float* W_out  = (const float*)d_in[15];
    const float* g_out  = (const float*)d_in[17];
    const float* be_out = (const float*)d_in[18];
    const float* attw   = (const float*)d_in[19];
    const float* attb   = (const float*)d_in[20];

    const int N = in_sizes[0] / HDIM;
    const int E = in_sizes[1] / 2;
    const size_t NH = (size_t)N * HDIM;
    const int Bg = (out_size - (int)NH) / HDIM;   // num graphs

    const int* src = ei;
    const int* dst = ei + E;

    // workspace layout (fp32 units)
    float* ws    = (float*)d_ws;
    float* hA    = ws;                 // N*128
    float* hB    = hA + NH;            // N*128
    float* hw    = hB + NH;            // N*128 (GEMM out)
    float* agg   = hw + NH;            // N*128 (scatter target)
    float* dinv  = agg + NH;           // N
    float* deg   = dinv + N;           // N
    float* sum   = deg + N;            // 128
    float* sumsq = sum + HDIM;         // 128
    _Float16* Wt = (_Float16*)(sumsq + HDIM);   // 4 * 128*128 fp16

    float* outNodes  = (float*)d_out;            // N*128
    float* outGraphs = outNodes + NH;            // Bg*128

    const int T = 256;

    // --- one-time prep -----------------------------------------------------
    wt_convert<<<cdiv(HDIM * HDIM, T), T, 0, stream>>>(W_in,  Wt + 0 * HDIM * HDIM);
    wt_convert<<<cdiv(HDIM * HDIM, T), T, 0, stream>>>(W_m0,  Wt + 1 * HDIM * HDIM);
    wt_convert<<<cdiv(HDIM * HDIM, T), T, 0, stream>>>(W_m1,  Wt + 2 * HDIM * HDIM);
    wt_convert<<<cdiv(HDIM * HDIM, T), T, 0, stream>>>(W_out, Wt + 3 * HDIM * HDIM);

    fill_f32<<<cdiv(N, T), T, 0, stream>>>(deg, 1.0f, N);            // self-loop
    deg_count<<<cdiv(E, T), T, 0, stream>>>(dst, deg, E);
    dinv_kernel<<<cdiv(N, T), T, 0, stream>>>(deg, dinv, N);

    // --- layers ------------------------------------------------------------
    const int nRowTiles = cdiv(N, 16);
    const int gemmBlocks = cdiv((long long)nRowTiles * 32, T);
    const int edgeBlocks = cdiv((long long)E * 32, T);
    const int elemBlocks = cdiv((long long)NH, T);
    const int statBlocks = cdiv(N, 64);

    struct Layer { const float* in; const _Float16* wt; const float* g; const float* be;
                   const float* resid; float* out; int relu; };
    Layer layers[4] = {
        { x,  Wt + 0 * HDIM * HDIM, g_in,  be_in,  nullptr, hA,       1 },
        { hA, Wt + 1 * HDIM * HDIM, g_m0,  be_m0,  hA,      hB,       1 },
        { hB, Wt + 2 * HDIM * HDIM, g_m1,  be_m1,  hB,      hA,       1 },
        { hA, Wt + 3 * HDIM * HDIM, g_out, be_out, nullptr, outNodes, 0 },
    };

    for (int l = 0; l < 4; ++l) {
        gemm128_wmma<<<gemmBlocks, T, 0, stream>>>(layers[l].in, layers[l].wt, hw, nRowTiles, N);
        fill_f32<<<elemBlocks, T, 0, stream>>>(agg, 0.0f, (int)NH);
        fill_f32<<<1, 2 * HDIM, 0, stream>>>(sum, 0.0f, 2 * HDIM);   // sum + sumsq
        edge_scatter<<<edgeBlocks, T, 0, stream>>>(hw, dinv, src, dst, agg, E);
        selfloop_stats<<<statBlocks, HDIM, 0, stream>>>(hw, dinv, agg, sum, sumsq, N);
        bn_apply<<<elemBlocks, T, 0, stream>>>(agg, sum, sumsq, layers[l].g, layers[l].be,
                                               layers[l].resid, layers[l].out, N, layers[l].relu);
    }

    // --- attention pooling -------------------------------------------------
    fill_f32<<<cdiv(Bg * HDIM, T), T, 0, stream>>>(outGraphs, 0.0f, Bg * HDIM);
    pool_attn<<<cdiv((long long)N * 32, T), T, 0, stream>>>(outNodes, attw, attb, batch,
                                                            outGraphs, N);
}
